// StandAttention_2319282340595
// MI455X (gfx1250) — matrix operand order repaired and checked by reading, not compiled, not run on hardware
//
#include <hip/hip_runtime.h>
#include <hip/hip_bf16.h>

// ---------------------------------------------------------------------------
// Single-head causal attention for MI455X (gfx1250), bf16 WMMA + f32 accum.
// B=4, S=4096, D=1024.  Flash-attention: scores never round-trip to HBM.
// CDNA5 paths used: v_wmma_f32_16x16x32_bf16, global_load_async_to_lds_b128
// (+ s_wait_asynccnt double buffering), global_prefetch_b8.
// Workspace layout (needs ~136MB):
//   [0..8MB)    WqT,WkT,WvT,WoT  (bf16, d x d each, transposed: [n][k])
//   [8..40MB)   Q   bf16 [b][s][d]   (pre-scaled by 1/sqrt(d))
//   [40..72MB)  K   bf16 [b][s][d]
//   [72..104MB) V^T bf16 [b][d][s]
//   [104..136MB)CTX bf16 [b][s][d]
// ---------------------------------------------------------------------------

#define DEV __device__ __forceinline__

static constexpr int B = 4;
static constexpr int S = 4096;
static constexpr int D = 1024;

typedef __attribute__((ext_vector_type(16))) __bf16 v16bf;
typedef __attribute__((ext_vector_type(8)))  __bf16 v8bf;
typedef __attribute__((ext_vector_type(8)))  float  v8f;

DEV v8f wmma_bf16(v16bf a, v16bf b, v8f c) {
  return __builtin_amdgcn_wmma_f32_16x16x32_bf16(
      /*neg_a=*/false, a, /*neg_b=*/false, b,
      /*c_mod=*/(short)0, c, /*reuse_a=*/false, /*reuse_b=*/false);
}

// --- A-matrix fragment (16x32, bf16), row-major source, lane layout per ISA:
// lanes 0-15 : row M=lane,    K = {k0..k0+7,  k0+16..k0+23}
// lanes 16-31: row M=lane-16, K = {k0+8..+15, k0+24..k0+31}
DEV v16bf load_a_bf16(const __bf16* tile, int ld, int k0) {
  int lane = threadIdx.x & 31;
  int half = lane >> 4;
  int m    = lane & 15;
  const __bf16* r = tile + (long)m * ld + k0 + half * 8;
  v8bf lo = *(const v8bf*)(r);
  v8bf hi = *(const v8bf*)(r + 16);
  return __builtin_shufflevector(lo, hi, 0,1,2,3,4,5,6,7,8,9,10,11,12,13,14,15);
}

DEV v16bf load_a_f32(const float* tile, int ld, int k0) {
  int lane = threadIdx.x & 31;
  int half = lane >> 4;
  int m    = lane & 15;
  const float* r = tile + (long)m * ld + k0 + half * 8;
  v16bf a;
#pragma unroll
  for (int e = 0; e < 8; ++e) {
    a[e]     = (__bf16)r[e];
    a[e + 8] = (__bf16)r[e + 16];
  }
  return a;
}

// --- B-matrix fragment (32x16 = KxN, bf16) from an [n][k] row-major source
// (i.e. the transposed weight / K-matrix / V^T).  Lane layout per ISA:
// lane n=lane&15, K = {k0+8h..+7, k0+16+8h..+7} (h = lane>=16) -> matches A layout.
DEV v16bf load_b_nk(const __bf16* src, long ld, int n0, long k0) {
  int lane = threadIdx.x & 31;
  int half = lane >> 4;
  int n    = lane & 15;
  const __bf16* r = src + (long)(n0 + n) * ld + k0 + half * 8;
  v8bf lo = *(const v8bf*)(r);
  v8bf hi = *(const v8bf*)(r + 16);
  return __builtin_shufflevector(lo, hi, 0,1,2,3,4,5,6,7,8,9,10,11,12,13,14,15);
}

// --- B-matrix fragment from an LDS-staged [rows][128] bf16 tile (ds_load_b128)
DEV v16bf load_b_lds(const __bf16* stage, int n0, int k0) {
  int lane = threadIdx.x & 31;
  int half = lane >> 4;
  int n    = lane & 15;
  const __bf16* r = stage + (n0 + n) * 128 + k0 + half * 8;
  v8bf lo = *(const v8bf*)(r);
  v8bf hi = *(const v8bf*)(r + 16);
  return __builtin_shufflevector(lo, hi, 0,1,2,3,4,5,6,7,8,9,10,11,12,13,14,15);
}

// ---------------------------------------------------------------------------
// Kernel 1: W (f32 [k][n]) -> W^T (bf16 [n][k])
// ---------------------------------------------------------------------------
__global__ void transpose_cast_kernel(const float* __restrict__ src,
                                      __bf16* __restrict__ dst) {
  for (long i = (long)blockIdx.x * blockDim.x + threadIdx.x;
       i < (long)D * D; i += (long)gridDim.x * blockDim.x) {
    int kk = (int)(i / D), nn = (int)(i % D);
    dst[(long)nn * D + kk] = (__bf16)src[i];
  }
}

// ---------------------------------------------------------------------------
// Kernel 2: fused QKV projection.  One wave per 16x16 output tile; one x
// A-fragment feeds 3 WMMAs.  Q scaled by 1/sqrt(D).  V stored transposed.
// jobs = (B*S/16)*(D/16) = 65536, 8 waves/block -> 8192 blocks.
// ---------------------------------------------------------------------------
__global__ void qkv_kernel(const float* __restrict__ x,
                           const __bf16* __restrict__ WqT,
                           const __bf16* __restrict__ WkT,
                           const __bf16* __restrict__ WvT,
                           const float* __restrict__ bq,
                           const float* __restrict__ bk,
                           const float* __restrict__ bv,
                           __bf16* __restrict__ q,
                           __bf16* __restrict__ k,
                           __bf16* __restrict__ vT) {
  const int lane = threadIdx.x & 31;
  const int wave = threadIdx.x >> 5;
  const long job = (long)blockIdx.x * 8 + wave;
  const int  n0   = (int)(job & 63) * 16;
  const long row0 = (job >> 6) * 16;        // global row in [0, B*S)
  const int  b    = (int)(row0 / S);
  const int  m0   = (int)(row0 % S);

  v8f accq = {0,0,0,0,0,0,0,0};
  v8f acck = {0,0,0,0,0,0,0,0};
  v8f accv = {0,0,0,0,0,0,0,0};
  const float* xr = x + row0 * D;
  for (int k0 = 0; k0 < D; k0 += 32) {
    v16bf a  = load_a_f32(xr, D, k0);
    accq = wmma_bf16(a, load_b_nk(WqT, D, n0, k0), accq);
    acck = wmma_bf16(a, load_b_nk(WkT, D, n0, k0), acck);
    accv = wmma_bf16(a, load_b_nk(WvT, D, n0, k0), accv);
  }

  const int half = lane >> 4, nn = lane & 15;
  const float scale = 0.03125f;             // 1/sqrt(1024)
  const float biq = bq[n0 + nn], bik = bk[n0 + nn], biv = bv[n0 + nn];
#pragma unroll
  for (int r = 0; r < 8; ++r) {
    int mrow = half * 8 + r;                // C layout: VGPR r -> row r / 8+r
    long grow = (long)b * S + m0 + mrow;
    q[grow * D + n0 + nn] = (__bf16)((accq[r] + biq) * scale);
    k[grow * D + n0 + nn] = (__bf16)(acck[r] + bik);
    vT[((long)b * D + n0 + nn) * S + m0 + mrow] = (__bf16)(accv[r] + biv);
  }
}

// ---------------------------------------------------------------------------
// Kernel 3: flash attention.  One block (8 waves) per 16-row query tile.
// Wave w owns d-columns [w*128, w*128+128):
//   - its Q slice (4 A-fragments) lives in registers for the whole loop
//   - its K slice per 32-key block is double-buffered in LDS via
//     global_load_async_to_lds_b128 (ASYNCcnt pipeline)
// Split-K over waves for S = Q K^T, LDS reduce, redundant online softmax,
// P x V from V^T (contiguous global, L2-resident).  grid.x = B*S/16 = 1024.
// ---------------------------------------------------------------------------
__global__ void flash_kernel(const __bf16* __restrict__ q,
                             const __bf16* __restrict__ kmat,
                             const __bf16* __restrict__ vT,
                             __bf16* __restrict__ ctx) {
  __shared__ __align__(16) __bf16 kstage[8][2][32 * 128]; // 128KB: per-wave dbl buf
  __shared__ float s_part[8 * 2 * 16 * 16];   // [wave][kt][qrow][key] 16KB
  __shared__ float s_full[16 * 33];           // reduced S, padded vs bank conflicts

  const int lane = threadIdx.x & 31;
  const int wave = threadIdx.x >> 5;
  const int half = lane >> 4;
  const int nn   = lane & 15;

  const int qt = blockIdx.x;                  // 0..1023
  const int b  = qt / (S / 16);
  const int q0 = (qt % (S / 16)) * 16;

  const __bf16* qb = q    + (long)b * S * D + (long)q0 * D;  // 16 x D tile
  const __bf16* kb = kmat + (long)b * S * D;
  const __bf16* vb = vT   + (long)b * D * S;
  const int dbase = wave * 128;               // this wave's 128 d-columns

  // ---- Q slice in registers for the whole key loop ----
  v16bf qa[4];
#pragma unroll
  for (int i = 0; i < 4; ++i) qa[i] = load_a_bf16(qb, D, dbase + i * 32);

  v8f acc[8];
#pragma unroll
  for (int t = 0; t < 8; ++t) acc[t] = (v8f){0,0,0,0,0,0,0,0};
  float mrow = -INFINITY;                     // per-lane state for row lane&15
  float lrow = 0.0f;

  const int kend    = q0 + 16;                // causal: keys < q0+16
  const int nblocks = (kend + 31) / 32;

  // async stage of one 32x128 bf16 K slice: 16 b128 copies per lane.
  // lane covers (row = half + 2i, 16B column chunk = lane&15).
  auto issue_stage = [&](int buf, int kb0) {
    asm volatile("s_wait_dscnt 0x0" ::: "memory"); // buffer reads done before refill
    const char* gbase = (const char*)(kb + (long)(kb0 + half) * D + dbase + nn * 8);
    unsigned lbase = (unsigned)(unsigned long long)(const void*)
                     &kstage[wave][buf][(half)*128 + nn * 8];
#pragma unroll
    for (int i = 0; i < 16; ++i) {
      unsigned long long ga = (unsigned long long)(gbase + (long)2 * i * D * 2);
      unsigned la = lbase + (unsigned)(2 * i * 128 * 2);
      asm volatile("global_load_async_to_lds_b128 %0, %1, off"
                   :: "v"(la), "v"(ga) : "memory");
    }
  };

  issue_stage(0, 0);

  for (int ib = 0; ib < nblocks; ++ib) {
    const int kb0 = ib * 32;
    const int buf = ib & 1;
    if (ib + 1 < nblocks) {
      issue_stage(buf ^ 1, kb0 + 32);
      // prefetch next V^T block lines into L2/L0 as well
      __builtin_prefetch(vb + (long)(dbase + lane) * S + kb0 + 32, 0, 1);
      asm volatile("s_wait_asynccnt 0x10" ::: "memory"); // current 16 done, next in flight
    } else {
      asm volatile("s_wait_asynccnt 0x0" ::: "memory");
    }
    const __bf16* kst = &kstage[wave][buf][0];

    // ---- partial S over this wave's 128-d slice: 2 key tiles of 16 ----
    v8f s0 = {0,0,0,0,0,0,0,0}, s1 = {0,0,0,0,0,0,0,0};
#pragma unroll
    for (int i = 0; i < 4; ++i) {
      s0 = wmma_bf16(qa[i], load_b_lds(kst, 0,  i * 32), s0);
      s1 = wmma_bf16(qa[i], load_b_lds(kst, 16, i * 32), s1);
    }
#pragma unroll
    for (int r = 0; r < 8; ++r) {
      int mr = half * 8 + r;
      s_part[wave * 512 +   0 + mr * 16 + nn] = s0[r];
      s_part[wave * 512 + 256 + mr * 16 + nn] = s1[r];
    }
    __syncthreads();

    // ---- reduce 8 partials -> full 16x32 S tile ----
    for (int e = threadIdx.x; e < 512; e += 256) {
      int kt = e >> 8, rr = (e >> 4) & 15, cc = e & 15;
      float sum = 0.0f;
#pragma unroll
      for (int w = 0; w < 8; ++w) sum += s_part[w * 512 + kt * 256 + rr * 16 + cc];
      s_full[rr * 33 + kt * 16 + cc] = sum;
    }
    __syncthreads();

    // ---- gather S in A-fragment layout + causal mask ----
    const int m = lane & 15;
    const int qrow = q0 + m;
    float sv[16];
#pragma unroll
    for (int e = 0; e < 16; ++e) {
      int kkidx = (e < 8 ? e : e + 8) + half * 8;   // K index within block
      int key = kb0 + kkidx;
      float v = s_full[m * 33 + kkidx];
      sv[e] = (key <= qrow) ? v : -INFINITY;
    }

    // ---- online softmax (redundant per wave, identical arithmetic) ----
    float bmax = sv[0];
#pragma unroll
    for (int e = 1; e < 16; ++e) bmax = fmaxf(bmax, sv[e]);
    bmax = fmaxf(bmax, __shfl_xor(bmax, 16, 32));   // join the two lane halves
    float mnew = fmaxf(mrow, bmax);
    float bsum = 0.0f;
#pragma unroll
    for (int e = 0; e < 16; ++e) { sv[e] = __expf(sv[e] - mnew); bsum += sv[e]; }
    bsum += __shfl_xor(bsum, 16, 32);
    float alpha = __expf(mrow - mnew);              // 0 on first block
    lrow = lrow * alpha + bsum;
    mrow = mnew;

    // rescale existing accumulators: alpha lives at lane == row
    float ar[8];
#pragma unroll
    for (int r = 0; r < 8; ++r) ar[r] = __shfl(alpha, half * 8 + r, 32);
#pragma unroll
    for (int t = 0; t < 8; ++t)
#pragma unroll
      for (int r = 0; r < 8; ++r) acc[t][r] *= ar[r];

    // ---- P (bf16 A-frag) x V : 8 n-tiles of this wave's d slice ----
    v16bf pa;
#pragma unroll
    for (int e = 0; e < 16; ++e) pa[e] = (__bf16)sv[e];
#pragma unroll
    for (int t = 0; t < 8; ++t) {
      v16bf bfrag = load_b_nk(vb, S, dbase + t * 16, kb0);  // V^T: [dcol][key]
      acc[t] = wmma_bf16(pa, bfrag, acc[t]);
    }
    __syncthreads();   // protect s_part/s_full before next iteration's writes
  }

  // ---- epilogue: divide by row sum, store ctx (bf16) ----
  float linv = 1.0f / lrow;
  float lr[8];
#pragma unroll
  for (int r = 0; r < 8; ++r) lr[r] = __shfl(linv, half * 8 + r, 32);
#pragma unroll
  for (int t = 0; t < 8; ++t)
#pragma unroll
    for (int r = 0; r < 8; ++r) {
      long grow = (long)b * S + q0 + half * 8 + r;
      ctx[grow * D + dbase + t * 16 + nn] = (__bf16)(acc[t][r] * lr[r]);
    }
}

// ---------------------------------------------------------------------------
// Kernel 4: out = ctx @ Wo + bo  (f32 output)
// ---------------------------------------------------------------------------
__global__ void proj_kernel(const __bf16* __restrict__ ctx,
                            const __bf16* __restrict__ WoT,
                            const float* __restrict__ bo,
                            float* __restrict__ out) {
  const int lane = threadIdx.x & 31;
  const int wave = threadIdx.x >> 5;
  const long job = (long)blockIdx.x * 8 + wave;
  const int  n0   = (int)(job & 63) * 16;
  const long row0 = (job >> 6) * 16;

  v8f acc = {0,0,0,0,0,0,0,0};
  const __bf16* cr = ctx + row0 * D;
  for (int k0 = 0; k0 < D; k0 += 32) {
    v16bf a = load_a_bf16(cr, D, k0);
    acc = wmma_bf16(a, load_b_nk(WoT, D, n0, k0), acc);
  }
  const int half = lane >> 4, nn = lane & 15;
  const float bias = bo[n0 + nn];
#pragma unroll
  for (int r = 0; r < 8; ++r)
    out[(row0 + half * 8 + r) * D + n0 + nn] = acc[r] + bias;
}

// ---------------------------------------------------------------------------
extern "C" void kernel_launch(void* const* d_in, const int* in_sizes, int n_in,
                              void* d_out, int out_size, void* d_ws, size_t ws_size,
                              hipStream_t stream) {
  (void)in_sizes; (void)n_in; (void)out_size; (void)ws_size;
  const float* x  = (const float*)d_in[0];
  const float* Wq = (const float*)d_in[1];
  const float* bq = (const float*)d_in[2];
  const float* Wk = (const float*)d_in[3];
  const float* bk = (const float*)d_in[4];
  const float* Wv = (const float*)d_in[5];
  const float* bv = (const float*)d_in[6];
  const float* Wo = (const float*)d_in[7];
  const float* bo = (const float*)d_in[8];
  float* out = (float*)d_out;

  char* p = (char*)d_ws;
  const size_t wmat = (size_t)D * D * sizeof(__bf16);       // 2 MB
  const size_t amat = (size_t)B * S * D * sizeof(__bf16);   // 32 MB
  __bf16* WqT = (__bf16*)p;             p += wmat;
  __bf16* WkT = (__bf16*)p;             p += wmat;
  __bf16* WvT = (__bf16*)p;             p += wmat;
  __bf16* WoT = (__bf16*)p;             p += wmat;
  __bf16* qbuf = (__bf16*)p;            p += amat;
  __bf16* kbuf = (__bf16*)p;            p += amat;
  __bf16* vTbuf = (__bf16*)p;           p += amat;
  __bf16* ctxbuf = (__bf16*)p;          p += amat;

  transpose_cast_kernel<<<1024, 256, 0, stream>>>(Wq, WqT);
  transpose_cast_kernel<<<1024, 256, 0, stream>>>(Wk, WkT);
  transpose_cast_kernel<<<1024, 256, 0, stream>>>(Wv, WvT);
  transpose_cast_kernel<<<1024, 256, 0, stream>>>(Wo, WoT);

  qkv_kernel<<<(B * S / 16) * (D / 16) / 8, 256, 0, stream>>>(
      x, WqT, WkT, WvT, bq, bk, bv, qbuf, kbuf, vTbuf);

  flash_kernel<<<B * S / 16, 256, 0, stream>>>(qbuf, kbuf, vTbuf, ctxbuf);

  proj_kernel<<<(B * S / 16) * (D / 16) / 8, 256, 0, stream>>>(
      ctxbuf, WoT, bo, out);
}
